// Classifier_z_only_13984413515933
// MI455X (gfx1250) — matrix operand order repaired and checked by reading, not compiled, not run on hardware
//
#include <hip/hip_runtime.h>
#include <hip/hip_bf16.h>
#include <math.h>

// ---------------------------------------------------------------------------
// Types for CDNA5 WMMA (gfx1250, wave32)
// ---------------------------------------------------------------------------
typedef __attribute__((ext_vector_type(16))) _Float16 v16h;
typedef __attribute__((ext_vector_type(8)))  _Float16 v8h;
typedef __attribute__((ext_vector_type(8)))  float    v8f;

#define EPSN  1e-5f
#define SLOPE 0.2f
#define NNODE 40000
#define NEDGE 320000

__device__ __forceinline__ v16h mk_frag(v8h lo, v8h hi) {
  v16h r;
#pragma unroll
  for (int i = 0; i < 8; ++i) { r[i] = lo[i]; r[8 + i] = hi[i]; }
  return r;
}
__device__ __forceinline__ float sigmf(float x) { return 1.0f / (1.0f + expf(-x)); }
// order-preserving float<->uint encoding for atomicMax on floats
__device__ __forceinline__ unsigned f2u_ord(float f) {
  unsigned u = __float_as_uint(f);
  return (u & 0x80000000u) ? ~u : (u | 0x80000000u);
}
__device__ __forceinline__ float u2f_ord(unsigned u) {
  return (u & 0x80000000u) ? __uint_as_float(u & 0x7FFFFFFFu) : __uint_as_float(~u);
}

// ---------------------------------------------------------------------------
// Utility kernels
// ---------------------------------------------------------------------------
__global__ void fill_u32_kernel(unsigned* __restrict__ p, unsigned v, long n) {
  long i = (long)blockIdx.x * blockDim.x + threadIdx.x;
  if (i < n) p[i] = v;
}

// f32 (sr,sc) row-major -> f16 (dr,dc), zero padded
__global__ void pad_f16_kernel(const float* __restrict__ src, _Float16* __restrict__ dst,
                               int sr, int sc, int dr, int dc) {
  long i = (long)blockIdx.x * blockDim.x + threadIdx.x;
  if (i >= (long)dr * dc) return;
  int r = (int)(i / dc), c = (int)(i % dc);
  dst[i] = (r < sr && c < sc) ? (_Float16)src[(size_t)r * sc + c] : (_Float16)0.f;
}

__global__ void pad_f32_kernel(const float* __restrict__ src, float* __restrict__ dst,
                               int sn, int dn) {
  int i = blockIdx.x * blockDim.x + threadIdx.x;
  if (i < dn) dst[i] = (i < sn) ? src[i] : 0.f;
}

// per-column mean / biased variance over M rows of an (M,N) f32 matrix; 1 block per column
__global__ void colstats_kernel(const float* __restrict__ X, int M, int N,
                                float* __restrict__ mean, float* __restrict__ var) {
  __shared__ float s1[256], s2[256];
  int j = blockIdx.x;
  float a = 0.f, b = 0.f;
  for (int i = threadIdx.x; i < M; i += blockDim.x) {
    float v = X[(size_t)i * N + j];
    a += v; b += v * v;
  }
  s1[threadIdx.x] = a; s2[threadIdx.x] = b;
  __syncthreads();
  for (int s = 128; s > 0; s >>= 1) {
    if ((int)threadIdx.x < s) { s1[threadIdx.x] += s1[threadIdx.x + s]; s2[threadIdx.x] += s2[threadIdx.x + s]; }
    __syncthreads();
  }
  if (threadIdx.x == 0) {
    float m = s1[0] / (float)M;
    mean[j] = m;
    var[j]  = s2[0] / (float)M - m * m;
  }
}

// BN apply (+optional ReLU); optional f32 dst (in-place OK) and/or zero-padded f16 dst (Mp,Np)
__global__ void bn_relu_dual_kernel(const float* __restrict__ src, int M, int N,
                                    const float* __restrict__ mean, const float* __restrict__ var,
                                    const float* __restrict__ g, const float* __restrict__ b,
                                    int relu, float* __restrict__ dst32,
                                    _Float16* __restrict__ dst16, int Mp, int Np) {
  long idx = (long)blockIdx.x * blockDim.x + threadIdx.x;
  if (idx >= (long)Mp * Np) return;
  int i = (int)(idx / Np), j = (int)(idx % Np);
  if (i < M && j < N) {
    float v = (src[(size_t)i * N + j] - mean[j]) * rsqrtf(var[j] + EPSN) * g[j] + b[j];
    if (relu) v = fmaxf(v, 0.f);
    if (dst32) dst32[(size_t)i * N + j] = v;
    if (dst16) dst16[idx] = (_Float16)v;
  } else if (dst16) {
    dst16[idx] = (_Float16)0.f;
  }
}

// ---------------------------------------------------------------------------
// WMMA GEMM:  C[M,N](f32) = A[M,Kp](f16, row-major) x W[N,Kp](f16, row-major)^T
// + bias (per col) + optional relu.  Kp % 32 == 0, M % 16 == 0, N % 16 == 0.
// One wave per 16x16 C-tile.  A fragment: lane holds row (tm+l16), K-chunk
// [k+16h, k+16h+15]; B fragment: lane holds W-row (tn+l16),
// K-chunk [k+16h, k+16h+15] (contiguous 32B).
// ---------------------------------------------------------------------------
__global__ void __launch_bounds__(256)
gemm_xwT_kernel(const _Float16* __restrict__ A, const _Float16* __restrict__ W,
                const float* __restrict__ bias, float* __restrict__ C,
                int M, int N, int Kp, int relu) {
  const int lane = threadIdx.x & 31;
  const int wid  = blockIdx.x * (blockDim.x >> 5) + (threadIdx.x >> 5);
  const int tilesN = N >> 4;
  const int totTiles = (M >> 4) * tilesN;
  if (wid >= totTiles) return;                       // wave-uniform exit
  const int tm  = (wid / tilesN) << 4;
  const int tn  = (wid % tilesN) << 4;
  const int hlf = lane >> 4, l16 = lane & 15;
  const _Float16* arow = A + (size_t)(tm + l16) * Kp;
  const _Float16* wrow = W + (size_t)(tn + l16) * Kp;
  v8f acc = {};
  for (int k = 0; k < Kp; k += 32) {
    __builtin_prefetch(arow + k + 128, 0, 0);        // global_prefetch_b8
    v16h fa = *(const v16h*)(arow + k + 16 * hlf);
    v16h fb = *(const v16h*)(wrow + k + 16 * hlf);
    acc = __builtin_amdgcn_wmma_f32_16x16x32_f16(false, fa, false, fb, (short)0, acc,
                                                 false, false);
  }
  const int row0 = tm + (hlf << 3);
  const int col  = tn + l16;
  float bv = bias ? bias[col] : 0.f;
#pragma unroll
  for (int r = 0; r < 8; ++r) {
    float v = acc[r] + bv;
    if (relu) v = fmaxf(v, 0.f);
    C[(size_t)(row0 + r) * N + col] = v;
  }
}

// ---------------------------------------------------------------------------
// Conv encoder helpers
// ---------------------------------------------------------------------------
// X (100,5,5,6,6) NCDHW f32 -> A (18000,160) f16 (K = 5*27 = 135, padded to 160)
__global__ void im2col1_kernel(const float* __restrict__ X, _Float16* __restrict__ A) {
  long idx = (long)blockIdx.x * blockDim.x + threadIdx.x;
  if (idx >= (long)18000 * 160) return;
  int row = (int)(idx / 160), k = (int)(idx % 160);
  _Float16 v = (_Float16)0.f;
  if (k < 135) {
    int c = k / 27, kk = k % 27;
    int kd = kk / 9, khy = (kk / 3) % 3, kwx = kk % 3;
    int f = row / 180, sp = row % 180;
    int d = sp / 36, hh = (sp / 6) % 6, ww = sp % 6;
    int z = d + kd - 1, y = hh + khy - 1, x = ww + kwx - 1;
    if (z >= 0 && z < 5 && y >= 0 && y < 6 && x >= 0 && x < 6)
      v = (_Float16)X[((((size_t)f * 5 + c) * 5 + z) * 6 + y) * 6 + x];
  }
  A[idx] = v;
}

// channels-last (F,D,H,W,C) f32 -> im2col A (F*D*H*W, Kp) f16, K = C*27, pad 1
__global__ void im2col_cl_kernel(const float* __restrict__ in, _Float16* __restrict__ A,
                                 int F, int D, int H, int W, int Cin, int Kp) {
  long idx = (long)blockIdx.x * blockDim.x + threadIdx.x;
  long total = (long)F * D * H * W * Kp;
  if (idx >= total) return;
  int row = (int)(idx / Kp), k = (int)(idx % Kp);
  int K = Cin * 27;
  _Float16 v = (_Float16)0.f;
  if (k < K) {
    int c = k / 27, kk = k % 27;
    int kd = kk / 9, khy = (kk / 3) % 3, kwx = kk % 3;
    int sp = row % (D * H * W), f = row / (D * H * W);
    int d = sp / (H * W), hh = (sp / W) % H, ww = sp % W;
    int z = d + kd - 1, y = hh + khy - 1, x = ww + kwx - 1;
    if (z >= 0 && z < D && y >= 0 && y < H && x >= 0 && x < W)
      v = (_Float16)in[((((size_t)f * D + z) * H + y) * W + x) * Cin + c];
  }
  A[idx] = v;
}

// channels-last maxpool, kernel 2 stride 2, -inf padding (pd,ph,pw)
__global__ void maxpool_cl_kernel(const float* __restrict__ in, float* __restrict__ out,
                                  int F, int D, int H, int W, int C,
                                  int Do, int Ho, int Wo, int pd, int ph, int pw) {
  long idx = (long)blockIdx.x * blockDim.x + threadIdx.x;
  if (idx >= (long)F * Do * Ho * Wo * C) return;
  int c = (int)(idx % C); long t = idx / C;
  int ow = (int)(t % Wo); t /= Wo;
  int oh = (int)(t % Ho); t /= Ho;
  int od = (int)(t % Do); int f = (int)(t / Do);
  float m = -3.4e38f;
  for (int dz = 0; dz < 2; ++dz) {
    int z = 2 * od - pd + dz; if (z < 0 || z >= D) continue;
    for (int dy = 0; dy < 2; ++dy) {
      int y = 2 * oh - ph + dy; if (y < 0 || y >= H) continue;
      for (int dx = 0; dx < 2; ++dx) {
        int x = 2 * ow - pw + dx; if (x < 0 || x >= W) continue;
        m = fmaxf(m, in[((((size_t)f * D + z) * H + y) * W + x) * C + c]);
      }
    }
  }
  out[idx] = m;
}

// maxpool (100,3,4,4,256) pads (1,0,0) + flatten to fc1 input order (C,D,H,W) as f16 (112,2048)
__global__ void pool2_flatten_kernel(const float* __restrict__ in, _Float16* __restrict__ A) {
  long idx = (long)blockIdx.x * blockDim.x + threadIdx.x;
  if (idx >= (long)112 * 2048) return;
  int f = (int)(idx / 2048), col = (int)(idx % 2048);
  if (f >= 100) { A[idx] = (_Float16)0.f; return; }
  int c = col >> 3, r = col & 7;
  int d = r >> 2, hh = (r >> 1) & 1, ww = r & 1;
  float m = -3.4e38f;
  for (int dz = 0; dz < 2; ++dz) {
    int z = 2 * d - 1 + dz; if (z < 0 || z >= 3) continue;
    for (int dy = 0; dy < 2; ++dy) {
      int y = 2 * hh + dy;
      for (int dx = 0; dx < 2; ++dx) {
        int x = 2 * ww + dx;
        m = fmaxf(m, in[((((size_t)f * 3 + z) * 4 + y) * 4 + x) * 256 + c]);
      }
    }
  }
  A[idx] = (_Float16)m;
}

// ---------------------------------------------------------------------------
// Persistent 2-layer-capable GRU step kernel: one workgroup (8 waves), 25 steps,
// h & x tiles in LDS, gates via WMMA (M-tile 16 padded from B=4, N=768, K=256)
// LDS: x16 (8K) | h16 (8K) | hf32 (16K) | gbuf 16x768 f32 (48K) | hn 16x256 f32 (16K)
// ---------------------------------------------------------------------------
__global__ void __launch_bounds__(256)
gru_layer_kernel(const _Float16* __restrict__ x, const _Float16* __restrict__ wih,
                 const _Float16* __restrict__ whh, const float* __restrict__ bih,
                 const float* __restrict__ bhh, _Float16* __restrict__ hout) {
  extern __shared__ char smem[];
  _Float16* x16 = (_Float16*)smem;                 // 16x256 f16
  _Float16* hh  = (_Float16*)(smem + 8192);        // 16x256 f16
  float*    hf  = (float*)(smem + 16384);          // 16x256 f32
  float*    gb  = (float*)(smem + 32768);          // 16x768 f32
  float*    hn  = (float*)(smem + 81920);          // 16x256 f32
  const int tid  = threadIdx.x;
  const int lane = tid & 31, wv = tid >> 5;
  const int hlf = lane >> 4, l16 = lane & 15;
  for (int i = tid; i < 16 * 256; i += 256) { hh[i] = (_Float16)0.f; hf[i] = 0.f; }
  __syncthreads();
  for (int t = 0; t < 25; ++t) {
    for (int i = tid; i < 16 * 256; i += 256) {
      int r = i >> 8, c = i & 255;
      x16[i] = (r < 4) ? x[(size_t)(r * 25 + t) * 256 + c] : (_Float16)0.f;
    }
    __syncthreads();
    for (int tt = wv; tt < 48; tt += 8) {
      const int tn = tt << 4;
      const _Float16* wi = wih + (size_t)(tn + l16) * 256;
      const _Float16* wh = whh + (size_t)(tn + l16) * 256;
      const _Float16* axr = x16 + l16 * 256;
      const _Float16* ahr = hh  + l16 * 256;
      v8f ai = {}; v8f ah = {};
#pragma unroll
      for (int k = 0; k < 256; k += 32) {
        v16h fa = *(const v16h*)(axr + k + 16 * hlf);
        v16h fh = *(const v16h*)(ahr + k + 16 * hlf);
        v16h bi = *(const v16h*)(wi + k + 16 * hlf);
        v16h bh = *(const v16h*)(wh + k + 16 * hlf);
        ai = __builtin_amdgcn_wmma_f32_16x16x32_f16(false, fa, false, bi, (short)0, ai, false, false);
        ah = __builtin_amdgcn_wmma_f32_16x16x32_f16(false, fh, false, bh, (short)0, ah, false, false);
      }
      const int row0 = hlf << 3, col = tn + l16;
      if (tn < 512) {
#pragma unroll
        for (int r = 0; r < 8; ++r) gb[(row0 + r) * 768 + col] = ai[r] + ah[r];
      } else {
#pragma unroll
        for (int r = 0; r < 8; ++r) {
          gb[(row0 + r) * 768 + col] = ai[r];
          hn[(row0 + r) * 256 + (col - 512)] = ah[r];
        }
      }
    }
    __syncthreads();
    for (int i = tid; i < 4 * 256; i += 256) {
      int r = i >> 8, c = i & 255;
      float rg = sigmf(gb[r * 768 + c] + bih[c] + bhh[c]);
      float zg = sigmf(gb[r * 768 + 256 + c] + bih[256 + c] + bhh[256 + c]);
      float ng = tanhf(gb[r * 768 + 512 + c] + bih[512 + c] + rg * (hn[r * 256 + c] + bhh[512 + c]));
      float hv = (1.f - zg) * ng + zg * hf[r * 256 + c];
      hf[r * 256 + c] = hv;
      hh[r * 256 + c] = (_Float16)hv;
      hout[(size_t)(r * 25 + t) * 256 + c] = (_Float16)hv;
    }
    __syncthreads();
  }
}

// h2 (rows b*25+t, 256) f16 -> dense A (16, 6400) f16, rows 4..15 zero
__global__ void pack_dense_kernel(const _Float16* __restrict__ h, _Float16* __restrict__ A) {
  long idx = (long)blockIdx.x * blockDim.x + threadIdx.x;
  if (idx >= (long)16 * 6400) return;
  int b = (int)(idx / 6400), k = (int)(idx % 6400);
  A[idx] = (b < 4) ? h[(size_t)(b * 25 + k / 256) * 256 + (k % 256)] : (_Float16)0.f;
}

// feats = bn(concat(node_x, enc[gid])) -> f16 (40000, 544) zero-padded past 513
__global__ void build_feats_kernel(const float* __restrict__ node_x, const float* __restrict__ enc,
                                   const float* __restrict__ mean, const float* __restrict__ var,
                                   const float* __restrict__ g, const float* __restrict__ b,
                                   _Float16* __restrict__ dst) {
  long idx = (long)blockIdx.x * blockDim.x + threadIdx.x;
  if (idx >= (long)NNODE * 544) return;
  int i = (int)(idx / 544), j = (int)(idx % 544);
  _Float16 v = (_Float16)0.f;
  if (j == 0) {
    v = (_Float16)((node_x[i] - mean[0]) * rsqrtf(var[0] + EPSN) * g[0] + b[0]);
  } else if (j < 513) {
    int gid = i / 10000;
    float xx = enc[(size_t)gid * 512 + (j - 1)];
    v = (_Float16)((xx - mean[j]) * rsqrtf(var[j] + EPSN) * g[j] + b[j]);
  }
  dst[idx] = v;
}

// ---------------------------------------------------------------------------
// GATv2 edge kernels
// ---------------------------------------------------------------------------
__global__ void edge_logits_kernel(const float* __restrict__ xl, const float* __restrict__ xr,
                                   const int* __restrict__ src, const int* __restrict__ dst,
                                   const float* __restrict__ att, int Hh, int C, int ldx,
                                   float* __restrict__ logits, unsigned* __restrict__ umax) {
  long i = (long)blockIdx.x * blockDim.x + threadIdx.x;
  if (i >= (long)NEDGE * Hh) return;
  int e = (int)(i / Hh), h = (int)(i % Hh);
  int s = src[e], d = dst[e];
  const float* pl = xl + (size_t)s * ldx + h * C;
  const float* pr = xr + (size_t)d * ldx + h * C;
  const float* pa = att + h * C;
  float acc = 0.f;
  for (int c = 0; c < C; ++c) {
    float v = pl[c] + pr[c];
    v = v > 0.f ? v : SLOPE * v;
    acc += pa[c] * v;
  }
  logits[i] = acc;
  atomicMax(&umax[(size_t)d * Hh + h], f2u_ord(acc));
}

__global__ void edge_exp_kernel(const int* __restrict__ dst, const unsigned* __restrict__ umax,
                                int Hh, float* __restrict__ logex, float* __restrict__ den) {
  long i = (long)blockIdx.x * blockDim.x + threadIdx.x;
  if (i >= (long)NEDGE * Hh) return;
  int e = (int)(i / Hh), h = (int)(i % Hh);
  int d = dst[e];
  float ex = expf(logex[i] - u2f_ord(umax[(size_t)d * Hh + h]));
  logex[i] = ex;
  atomicAdd(&den[(size_t)d * Hh + h], ex);
}

__global__ void edge_msg_kernel(const float* __restrict__ xl, const int* __restrict__ src,
                                const int* __restrict__ dst, const float* __restrict__ ex,
                                const float* __restrict__ den, int Hh, int C, int ldx,
                                float* __restrict__ out) {
  long i = (long)blockIdx.x * blockDim.x + threadIdx.x;
  if (i >= (long)NEDGE * Hh) return;
  int e = (int)(i / Hh), h = (int)(i % Hh);
  int s = src[e], d = dst[e];
  float alpha = ex[i] / (den[(size_t)d * Hh + h] + 1e-16f);
  const float* pl = xl + (size_t)s * ldx + h * C;
  float* po = out + (size_t)d * (Hh * C) + h * C;
  for (int c = 0; c < C; ++c) atomicAdd(&po[c], pl[c] * alpha);
}

__global__ void count_edges_kernel(const int* __restrict__ dst, float* __restrict__ cnt) {
  int e = blockIdx.x * blockDim.x + threadIdx.x;
  if (e < NEDGE) atomicAdd(&cnt[dst[e]], 1.f);
}

__global__ void gat_finalize_kernel(float* __restrict__ out, const float* __restrict__ cnt,
                                    const float* __restrict__ bias, int NC) {
  long i = (long)blockIdx.x * blockDim.x + threadIdx.x;
  if (i >= (long)NNODE * NC) return;
  int n = (int)(i / NC), j = (int)(i % NC);
  out[i] = out[i] / fmaxf(cnt[n], 1.f) + bias[j];
}

__global__ void final_out_kernel(const float* __restrict__ o, const unsigned char* __restrict__ mask,
                                 const float* __restrict__ y, float* __restrict__ dout) {
  int i = blockIdx.x * blockDim.x + threadIdx.x;
  if (i >= NNODE) return;
  float p = 1.f / (1.f + expf(-o[i]));
  dout[i] = mask[i] ? p : 0.f;
  dout[NNODE + i] = y[i];
}

// ---------------------------------------------------------------------------
// Host-side orchestration
// ---------------------------------------------------------------------------
static inline unsigned NB(long n) { return (unsigned)((n + 255) / 256); }

extern "C" void kernel_launch(void* const* d_in, const int* in_sizes, int n_in,
                              void* d_out, int out_size, void* d_ws, size_t ws_size,
                              hipStream_t stream) {
  (void)in_sizes; (void)n_in; (void)out_size; (void)ws_size;
  const float* X       = (const float*)d_in[0];
  const float* node_x  = (const float*)d_in[1];
  const float* Yv      = (const float*)d_in[2];
  const float* conv1_w = (const float*)d_in[3];  const float* conv1_b = (const float*)d_in[4];
  const float* bn1_g   = (const float*)d_in[5];  const float* bn1_b   = (const float*)d_in[6];
  const float* conv2_w = (const float*)d_in[7];  const float* conv2_b = (const float*)d_in[8];
  const float* bn2_g   = (const float*)d_in[9];  const float* bn2_b   = (const float*)d_in[10];
  const float* conv3_w = (const float*)d_in[11]; const float* conv3_b = (const float*)d_in[12];
  const float* bn3_g   = (const float*)d_in[13]; const float* bn3_b   = (const float*)d_in[14];
  const float* fc1_w   = (const float*)d_in[15]; const float* fc1_b   = (const float*)d_in[16];
  const float* bnf1_g  = (const float*)d_in[17]; const float* bnf1_b  = (const float*)d_in[18];
  const float* fc2_w   = (const float*)d_in[19]; const float* fc2_b   = (const float*)d_in[20];
  const float* bnf2_g  = (const float*)d_in[21]; const float* bnf2_b  = (const float*)d_in[22];
  const float* g0_wih  = (const float*)d_in[23]; const float* g0_whh  = (const float*)d_in[24];
  const float* g0_bih  = (const float*)d_in[25]; const float* g0_bhh  = (const float*)d_in[26];
  const float* g1_wih  = (const float*)d_in[27]; const float* g1_whh  = (const float*)d_in[28];
  const float* g1_bih  = (const float*)d_in[29]; const float* g1_bhh  = (const float*)d_in[30];
  const float* dense_w = (const float*)d_in[31]; const float* dense_b = (const float*)d_in[32];
  const float* gbn0_g  = (const float*)d_in[33]; const float* gbn0_b  = (const float*)d_in[34];
  const float* gat1_wl = (const float*)d_in[35]; const float* gat1_wr = (const float*)d_in[36];
  const float* gat1_bl = (const float*)d_in[37]; const float* gat1_br = (const float*)d_in[38];
  const float* gat1_at = (const float*)d_in[39]; const float* gat1_bs = (const float*)d_in[40];
  const float* gbn1_g  = (const float*)d_in[41]; const float* gbn1_b  = (const float*)d_in[42];
  const float* gat2_wl = (const float*)d_in[43]; const float* gat2_wr = (const float*)d_in[44];
  const float* gat2_bl = (const float*)d_in[45]; const float* gat2_br = (const float*)d_in[46];
  const float* gat2_at = (const float*)d_in[47]; const float* gat2_bs = (const float*)d_in[48];
  const float* gbn2_g  = (const float*)d_in[49]; const float* gbn2_b  = (const float*)d_in[50];
  const float* gat3_wl = (const float*)d_in[51]; const float* gat3_wr = (const float*)d_in[52];
  const float* gat3_bl = (const float*)d_in[53]; const float* gat3_br = (const float*)d_in[54];
  const float* gat3_at = (const float*)d_in[55]; const float* gat3_bs = (const float*)d_in[56];
  const int*   eidx    = (const int*)d_in[57];
  const unsigned char* mask = (const unsigned char*)d_in[58];
  const int* esrc = eidx;
  const int* edst = eidx + NEDGE;
  float* dout = (float*)d_out;

  // ---- workspace bump allocator ----
  char* wsb = (char*)d_ws; size_t off = 0;
  auto alloc = [&](size_t bytes) -> void* {
    void* p = wsb + off; off += (bytes + 255) & ~(size_t)255; return p;
  };
  _Float16* w1p   = (_Float16*)alloc((size_t)64 * 160 * 2);
  _Float16* w2p   = (_Float16*)alloc((size_t)64 * 1728 * 2);
  _Float16* w3p   = (_Float16*)alloc((size_t)256 * 1728 * 2);
  _Float16* fc1p  = (_Float16*)alloc((size_t)512 * 2048 * 2);
  _Float16* fc2p  = (_Float16*)alloc((size_t)256 * 512 * 2);
  _Float16* wih0p = (_Float16*)alloc((size_t)768 * 256 * 2);
  _Float16* whh0p = (_Float16*)alloc((size_t)768 * 256 * 2);
  _Float16* wih1p = (_Float16*)alloc((size_t)768 * 256 * 2);
  _Float16* whh1p = (_Float16*)alloc((size_t)768 * 256 * 2);
  _Float16* denp  = (_Float16*)alloc((size_t)512 * 6400 * 2);
  _Float16* w1l   = (_Float16*)alloc((size_t)256 * 544 * 2);
  _Float16* w1r   = (_Float16*)alloc((size_t)256 * 544 * 2);
  _Float16* w2l   = (_Float16*)alloc((size_t)128 * 256 * 2);
  _Float16* w2r   = (_Float16*)alloc((size_t)128 * 256 * 2);
  _Float16* w3l   = (_Float16*)alloc((size_t)16 * 128 * 2);
  _Float16* w3r   = (_Float16*)alloc((size_t)16 * 128 * 2);
  float* b3lp     = (float*)alloc(16 * 4);
  float* b3rp     = (float*)alloc(16 * 4);
  float* meanG    = (float*)alloc(1024 * 4);
  float* varG     = (float*)alloc(1024 * 4);
  _Float16* Acol  = (_Float16*)alloc((size_t)18000 * 1728 * 2);
  float* convout  = (float*)alloc((size_t)4800 * 256 * 4 > (size_t)18000 * 64 * 4 ?
                                  (size_t)4800 * 256 * 4 : (size_t)18000 * 64 * 4);
  float* pooled1  = (float*)alloc((size_t)100 * 48 * 64 * 4);
  _Float16* Afc1  = (_Float16*)alloc((size_t)112 * 2048 * 2);
  float* fcout    = (float*)alloc((size_t)112 * 512 * 4);
  _Float16* Afc2  = (_Float16*)alloc((size_t)112 * 512 * 2);
  float* fc2out   = (float*)alloc((size_t)112 * 256 * 4);
  _Float16* xg    = (_Float16*)alloc((size_t)112 * 256 * 2);
  _Float16* h1    = (_Float16*)alloc((size_t)100 * 256 * 2);
  _Float16* h2    = (_Float16*)alloc((size_t)100 * 256 * 2);
  _Float16* Aden  = (_Float16*)alloc((size_t)16 * 6400 * 2);
  float* enc      = (float*)alloc((size_t)16 * 512 * 4);
  _Float16* feats = (_Float16*)alloc((size_t)NNODE * 544 * 2);
  float* xlb      = (float*)alloc((size_t)NNODE * 256 * 4);
  float* xrb      = (float*)alloc((size_t)NNODE * 256 * 4);
  float* outacc   = (float*)alloc((size_t)NNODE * 256 * 4);
  float* elog     = (float*)alloc((size_t)NEDGE * 2 * 4);
  unsigned* umax  = (unsigned*)alloc((size_t)NNODE * 2 * 4);
  float* denb     = (float*)alloc((size_t)NNODE * 2 * 4);
  float* cntb     = (float*)alloc((size_t)NNODE * 4);

  auto gemm = [&](const _Float16* A, const _Float16* W, const float* bias, float* C,
                  int M, int N, int Kp, int relu) {
    long tiles = (long)(M / 16) * (N / 16);
    gemm_xwT_kernel<<<(unsigned)((tiles + 7) / 8), 256, 0, stream>>>(A, W, bias, C, M, N, Kp, relu);
  };

  // ---- weight conversions to padded f16 ----
  pad_f16_kernel<<<NB(64L * 160), 256, 0, stream>>>(conv1_w, w1p, 64, 135, 64, 160);
  pad_f16_kernel<<<NB(64L * 1728), 256, 0, stream>>>(conv2_w, w2p, 64, 1728, 64, 1728);
  pad_f16_kernel<<<NB(256L * 1728), 256, 0, stream>>>(conv3_w, w3p, 256, 1728, 256, 1728);
  pad_f16_kernel<<<NB(512L * 2048), 256, 0, stream>>>(fc1_w, fc1p, 512, 2048, 512, 2048);
  pad_f16_kernel<<<NB(256L * 512), 256, 0, stream>>>(fc2_w, fc2p, 256, 512, 256, 512);
  pad_f16_kernel<<<NB(768L * 256), 256, 0, stream>>>(g0_wih, wih0p, 768, 256, 768, 256);
  pad_f16_kernel<<<NB(768L * 256), 256, 0, stream>>>(g0_whh, whh0p, 768, 256, 768, 256);
  pad_f16_kernel<<<NB(768L * 256), 256, 0, stream>>>(g1_wih, wih1p, 768, 256, 768, 256);
  pad_f16_kernel<<<NB(768L * 256), 256, 0, stream>>>(g1_whh, whh1p, 768, 256, 768, 256);
  pad_f16_kernel<<<NB(512L * 6400), 256, 0, stream>>>(dense_w, denp, 512, 6400, 512, 6400);
  pad_f16_kernel<<<NB(256L * 544), 256, 0, stream>>>(gat1_wl, w1l, 256, 513, 256, 544);
  pad_f16_kernel<<<NB(256L * 544), 256, 0, stream>>>(gat1_wr, w1r, 256, 513, 256, 544);
  pad_f16_kernel<<<NB(128L * 256), 256, 0, stream>>>(gat2_wl, w2l, 128, 256, 128, 256);
  pad_f16_kernel<<<NB(128L * 256), 256, 0, stream>>>(gat2_wr, w2r, 128, 256, 128, 256);
  pad_f16_kernel<<<NB(16L * 128), 256, 0, stream>>>(gat3_wl, w3l, 1, 128, 16, 128);
  pad_f16_kernel<<<NB(16L * 128), 256, 0, stream>>>(gat3_wr, w3r, 1, 128, 16, 128);
  pad_f32_kernel<<<1, 32, 0, stream>>>(gat3_bl, b3lp, 1, 16);
  pad_f32_kernel<<<1, 32, 0, stream>>>(gat3_br, b3rp, 1, 16);

  // ---- Conv3D encoder (im2col + WMMA GEMM, channels-last activations) ----
  im2col1_kernel<<<NB(18000L * 160), 256, 0, stream>>>(X, Acol);
  gemm(Acol, w1p, conv1_b, convout, 18000, 64, 160, 0);
  colstats_kernel<<<64, 256, 0, stream>>>(convout, 18000, 64, meanG, varG);
  bn_relu_dual_kernel<<<NB(18000L * 64), 256, 0, stream>>>(convout, 18000, 64, meanG, varG,
      bn1_g, bn1_b, 1, convout, nullptr, 18000, 64);
  im2col_cl_kernel<<<NB(18000L * 1728), 256, 0, stream>>>(convout, Acol, 100, 5, 6, 6, 64, 1728);
  gemm(Acol, w2p, conv2_b, convout, 18000, 64, 1728, 0);
  colstats_kernel<<<64, 256, 0, stream>>>(convout, 18000, 64, meanG, varG);
  bn_relu_dual_kernel<<<NB(18000L * 64), 256, 0, stream>>>(convout, 18000, 64, meanG, varG,
      bn2_g, bn2_b, 1, convout, nullptr, 18000, 64);
  maxpool_cl_kernel<<<NB(100L * 48 * 64), 256, 0, stream>>>(convout, pooled1, 100, 5, 6, 6, 64,
      3, 4, 4, 1, 1, 1);
  im2col_cl_kernel<<<NB(4800L * 1728), 256, 0, stream>>>(pooled1, Acol, 100, 3, 4, 4, 64, 1728);
  gemm(Acol, w3p, conv3_b, convout, 4800, 256, 1728, 0);
  colstats_kernel<<<256, 256, 0, stream>>>(convout, 4800, 256, meanG, varG);
  bn_relu_dual_kernel<<<NB(4800L * 256), 256, 0, stream>>>(convout, 4800, 256, meanG, varG,
      bn3_g, bn3_b, 1, convout, nullptr, 4800, 256);
  pool2_flatten_kernel<<<NB(112L * 2048), 256, 0, stream>>>(convout, Afc1);

  // ---- FC head ----
  gemm(Afc1, fc1p, fc1_b, fcout, 112, 512, 2048, 0);
  colstats_kernel<<<512, 256, 0, stream>>>(fcout, 100, 512, meanG, varG);
  bn_relu_dual_kernel<<<NB(112L * 512), 256, 0, stream>>>(fcout, 100, 512, meanG, varG,
      bnf1_g, bnf1_b, 1, nullptr, Afc2, 112, 512);
  gemm(Afc2, fc2p, fc2_b, fc2out, 112, 256, 512, 0);
  colstats_kernel<<<256, 256, 0, stream>>>(fc2out, 100, 256, meanG, varG);
  bn_relu_dual_kernel<<<NB(112L * 256), 256, 0, stream>>>(fc2out, 100, 256, meanG, varG,
      bnf2_g, bnf2_b, 1, nullptr, xg, 112, 256);

  // ---- 2-layer GRU (persistent workgroup, WMMA in LDS) ----
  gru_layer_kernel<<<1, 256, 98304, stream>>>(xg, wih0p, whh0p, g0_bih, g0_bhh, h1);
  gru_layer_kernel<<<1, 256, 98304, stream>>>(h1, wih1p, whh1p, g1_bih, g1_bhh, h2);
  pack_dense_kernel<<<NB(16L * 6400), 256, 0, stream>>>(h2, Aden);
  gemm(Aden, denp, dense_b, enc, 16, 512, 6400, 1);

  // ---- gbn0 stats (col0 from node_x; cols 1..512 from enc since encoding repeats/graph) ----
  colstats_kernel<<<1, 256, 0, stream>>>(node_x, NNODE, 1, meanG, varG);
  colstats_kernel<<<512, 256, 0, stream>>>(enc, 4, 512, meanG + 1, varG + 1);
  build_feats_kernel<<<NB((long)NNODE * 544), 256, 0, stream>>>(node_x, enc, meanG, varG,
      gbn0_g, gbn0_b, feats);

  // ---- GATv2 stack ----
  auto gat = [&](const _Float16* xin, int Kp, const _Float16* wl, const _Float16* wr,
                 const float* bl, const float* br, const float* att, const float* bias,
                 int Hh, int C, int Np, float* outb) {
    int NC = Hh * C;
    gemm(xin, wl, bl, xlb, NNODE, Np, Kp, 0);
    gemm(xin, wr, br, xrb, NNODE, Np, Kp, 0);
    fill_u32_kernel<<<NB((long)NNODE * Hh), 256, 0, stream>>>(umax, 0x007FFFFFu, (long)NNODE * Hh);
    fill_u32_kernel<<<NB((long)NNODE * Hh), 256, 0, stream>>>((unsigned*)denb, 0u, (long)NNODE * Hh);
    fill_u32_kernel<<<NB((long)NNODE), 256, 0, stream>>>((unsigned*)cntb, 0u, (long)NNODE);
    fill_u32_kernel<<<NB((long)NNODE * NC), 256, 0, stream>>>((unsigned*)outb, 0u, (long)NNODE * NC);
    edge_logits_kernel<<<NB((long)NEDGE * Hh), 256, 0, stream>>>(xlb, xrb, esrc, edst, att,
        Hh, C, Np, elog, umax);
    edge_exp_kernel<<<NB((long)NEDGE * Hh), 256, 0, stream>>>(edst, umax, Hh, elog, denb);
    edge_msg_kernel<<<NB((long)NEDGE * Hh), 256, 0, stream>>>(xlb, esrc, edst, elog, denb,
        Hh, C, Np, outb);
    count_edges_kernel<<<NB((long)NEDGE), 256, 0, stream>>>(edst, cntb);
    gat_finalize_kernel<<<NB((long)NNODE * NC), 256, 0, stream>>>(outb, cntb, bias, NC);
  };

  gat(feats, 544, w1l, w1r, gat1_bl, gat1_br, gat1_at, gat1_bs, 2, 128, 256, outacc);
  colstats_kernel<<<256, 256, 0, stream>>>(outacc, NNODE, 256, meanG, varG);
  bn_relu_dual_kernel<<<NB((long)NNODE * 256), 256, 0, stream>>>(outacc, NNODE, 256, meanG, varG,
      gbn1_g, gbn1_b, 1, nullptr, feats, NNODE, 256);

  gat(feats, 256, w2l, w2r, gat2_bl, gat2_br, gat2_at, gat2_bs, 1, 128, 128, outacc);
  colstats_kernel<<<128, 256, 0, stream>>>(outacc, NNODE, 128, meanG, varG);
  bn_relu_dual_kernel<<<NB((long)NNODE * 128), 256, 0, stream>>>(outacc, NNODE, 128, meanG, varG,
      gbn2_g, gbn2_b, 1, nullptr, feats, NNODE, 128);

  gat(feats, 128, w3l, w3r, b3lp, b3rp, gat3_at, gat3_bs, 1, 1, 16, outacc);
  final_out_kernel<<<NB((long)NNODE), 256, 0, stream>>>(outacc, mask, Yv, dout);
}
